// VectorQuantizer_37383395344398
// MI455X (gfx1250) — compile-verified, matrix operand order checked by reading, not run on hardware
//
#include <hip/hip_runtime.h>

typedef __attribute__((ext_vector_type(16))) __bf16 v16bf;
typedef __attribute__((ext_vector_type(8)))  __bf16 v8bf;
typedef __attribute__((ext_vector_type(4)))  __bf16 v4bf;
typedef __attribute__((ext_vector_type(8)))  float  v8f;

#define N_TOK 8192
#define K_CB  8192
#define DIM   512
#define LDSS  136   // 128 + 8 bf16 pad -> 16B (4-bank) shift per row, conflict-free b128 reads

#ifndef __has_builtin
#define __has_builtin(x) 0
#endif
#if __has_builtin(__builtin_amdgcn_tensor_load_to_lds) && \
    __has_builtin(__builtin_amdgcn_s_wait_tensorcnt)
#define HAVE_TDM 1
#else
#define HAVE_TDM 0
#endif

union FragBF {
  v16bf v;
  struct { v8bf lo; v8bf hi; } p;
};

// ---------------------------------------------------------------------------
// Kernel 1: codebook squared norms, one wave per row
// ---------------------------------------------------------------------------
__global__ void vq_cnorm_kernel(const float* __restrict__ codebook,
                                float* __restrict__ cnorm) {
  const int w = threadIdx.x >> 5, l = threadIdx.x & 31;
  const int row = blockIdx.x * 8 + w;
  const float* c = codebook + (size_t)row * DIM;
  float s = 0.f;
  for (int d = l; d < DIM; d += 32) { float x = c[d]; s += x * x; }
  #pragma unroll
  for (int m = 16; m; m >>= 1) s += __shfl_xor(s, m, 32);
  if (l == 0) cnorm[row] = s;
}

// ---------------------------------------------------------------------------
// Kernel 1b: one-time fp32 -> split bf16 (hi + lo) pre-pass, one float4/thread
// ---------------------------------------------------------------------------
__global__ void vq_split_kernel(const float* __restrict__ src,
                                __bf16* __restrict__ hi,
                                __bf16* __restrict__ lo) {
  const int i = blockIdx.x * blockDim.x + threadIdx.x;      // float4 index
  const float4 v = ((const float4*)src)[i];
  float fx[4] = { v.x, v.y, v.z, v.w };
  v4bf h, lw;
  #pragma unroll
  for (int j = 0; j < 4; ++j) {
    const __bf16 hb = (__bf16)fx[j];
    h[j]  = hb;
    lw[j] = (__bf16)(fx[j] - (float)hb);
  }
  ((v4bf*)hi)[i] = h;
  ((v4bf*)lo)[i] = lw;
}

// ---------------------------------------------------------------------------
// TDM: DMA one 128x128 bf16 tile (global row stride 512 elems) into LDS with
// hardware padding producing the 136-element row stride (pad 16B per 256B row)
// ---------------------------------------------------------------------------
#if HAVE_TDM
typedef __attribute__((ext_vector_type(4))) unsigned int v4u;
typedef __attribute__((ext_vector_type(4))) int v4i;
typedef __attribute__((ext_vector_type(8))) int v8i;

__device__ __forceinline__ void vq_tdm_tile128(const __bf16* gsrc, unsigned int ldsOff) {
  const unsigned long long ga = (unsigned long long)(uintptr_t)gsrc;
  // D# group0: count=1 | lds_addr | global_addr[56:0] | type=2
  v4u g0 = { 1u, ldsOff, (unsigned int)ga,
             (unsigned int)(ga >> 32) | (2u << 30) };
  // D# group1: data_size=2B, pad_enable, pad_interval=64 DW (256B),
  // pad_amount=4 DW (16B); tensor_dim0=512, tensor_dim1=65535 (no OOB),
  // tile 128x128, tensor_dim0_stride=512 elems
  v8i g1 = {
    (int)((1u << 16) | (1u << 20) | (5u << 22) | (3u << 25)),
    (int)(512u << 16),          // tensor_dim0[15:0]
    (int)(0xFFFFu << 16),       // tensor_dim0 hi=0 | tensor_dim1[15:0]
    (int)(128u << 16),          // tensor_dim1 hi=0 | tile_dim0=128
    (int)128u,                  // tile_dim1=128, tile_dim2=0
    (int)512u,                  // tensor_dim0_stride[31:0]
    0, 0
  };
  v4i z4 = { 0, 0, 0, 0 };
#if __clang_major__ >= 23
  v8i z8 = { 0, 0, 0, 0, 0, 0, 0, 0 };
  __builtin_amdgcn_tensor_load_to_lds(g0, g1, z4, z4, z8, 0);
#else
  __builtin_amdgcn_tensor_load_to_lds(g0, g1, z4, z4, 0);
#endif
}
#else
// Fallback: manual bf16 tile copy (global, row stride DIM) -> padded LDS
__device__ __forceinline__ void vq_copy_tile_bf16(const __bf16* __restrict__ src,
                                                  __bf16* dst, int tid) {
  #pragma unroll
  for (int i = 0; i < 8; ++i) {
    const int v8 = tid + i * 256;     // 2048 16-byte chunks
    const int r  = v8 >> 4;
    const int c  = (v8 & 15) << 3;
    *(v8bf*)(dst + r * LDSS + c) = *(const v8bf*)(src + (size_t)r * DIM + c);
  }
}
#endif

// ---------------------------------------------------------------------------
// Shared WMMA inner chunk: 128 rows x 128 cols x 128 D, split-bf16 3-pass
// ---------------------------------------------------------------------------
__device__ __forceinline__ void vq_mma_chunk(const __bf16* Ah, const __bf16* Al,
                                             const __bf16* Bh, const __bf16* Bl,
                                             int w, int lcol, int lhalf,
                                             v8f acc[8]) {
  #pragma unroll
  for (int di = 0; di < 128; di += 32) {
    const int arow = (16 * w + lcol) * LDSS;
    const int k0   = di + 8 * lhalf;
    FragBF ah, al;
    ah.p.lo = *(const v8bf*)(Ah + arow + k0);
    ah.p.hi = *(const v8bf*)(Ah + arow + k0 + 16);
    al.p.lo = *(const v8bf*)(Al + arow + k0);
    al.p.hi = *(const v8bf*)(Al + arow + k0 + 16);
    #pragma unroll
    for (int t = 0; t < 8; ++t) {
      const int brow = (16 * t + lcol) * LDSS;
      FragBF bh, bl;
      bh.p.lo = *(const v8bf*)(Bh + brow + k0);
      bh.p.hi = *(const v8bf*)(Bh + brow + k0 + 16);
      bl.p.lo = *(const v8bf*)(Bl + brow + k0);
      bl.p.hi = *(const v8bf*)(Bl + brow + k0 + 16);
      acc[t] = __builtin_amdgcn_wmma_f32_16x16x32_bf16(
                   false, ah.v, false, bh.v, (short)0, acc[t], false, false);
      acc[t] = __builtin_amdgcn_wmma_f32_16x16x32_bf16(
                   false, ah.v, false, bl.v, (short)0, acc[t], false, false);
      acc[t] = __builtin_amdgcn_wmma_f32_16x16x32_bf16(
                   false, al.v, false, bh.v, (short)0, acc[t], false, false);
    }
  }
}

// ---------------------------------------------------------------------------
// Kernel 2a (preferred): GEMM + streaming argmin from pre-split bf16 arrays,
// tiles staged by the Tensor Data Mover. grid = (64 row-tiles, 8 K-splits)
// ---------------------------------------------------------------------------
__launch_bounds__(256)
__global__ void vq_dist_argmin_bf16_kernel(const __bf16* __restrict__ Lh,
                                           const __bf16* __restrict__ Ll,
                                           const __bf16* __restrict__ Ch,
                                           const __bf16* __restrict__ Cl,
                                           const float* __restrict__ cnorm,
                                           float* __restrict__ pval,
                                           int*   __restrict__ pidx) {
  __shared__ __bf16 Ah[128 * LDSS];
  __shared__ __bf16 Al[128 * LDSS];
  __shared__ __bf16 Bh[128 * LDSS];
  __shared__ __bf16 Bl[128 * LDSS];

  const int tid   = threadIdx.x;
  const int w     = tid >> 5;
  const int l     = tid & 31;
  const int lhalf = l >> 4;
  const int lcol  = l & 15;
  const int rowBase    = blockIdx.x * 128;
  const int kSplitBase = blockIdx.y * 1024;

  float bestv[8];
  int   besti[8];
  #pragma unroll
  for (int r = 0; r < 8; ++r) { bestv[r] = 3.0e38f; besti[r] = 0; }

  const v8f vzero = {0.f, 0.f, 0.f, 0.f, 0.f, 0.f, 0.f, 0.f};

  for (int kt = 0; kt < 8; ++kt) {
    const int kBase = kSplitBase + kt * 128;
    v8f acc[8];
    #pragma unroll
    for (int t = 0; t < 8; ++t) acc[t] = vzero;

    for (int dc = 0; dc < 4; ++dc) {
      __syncthreads();   // previous chunk's LDS reads complete
#if HAVE_TDM
      if (w == 0) {
        const size_t aoff = (size_t)rowBase * DIM + dc * 128;
        const size_t boff = (size_t)kBase   * DIM + dc * 128;
        vq_tdm_tile128(Lh + aoff, (unsigned int)(uintptr_t)Ah);
        vq_tdm_tile128(Ll + aoff, (unsigned int)(uintptr_t)Al);
        vq_tdm_tile128(Ch + boff, (unsigned int)(uintptr_t)Bh);
        vq_tdm_tile128(Cl + boff, (unsigned int)(uintptr_t)Bl);
        __builtin_amdgcn_s_wait_tensorcnt((short)0);
      }
#else
      {
        const size_t aoff = (size_t)rowBase * DIM + dc * 128;
        const size_t boff = (size_t)kBase   * DIM + dc * 128;
        vq_copy_tile_bf16(Lh + aoff, Ah, tid);
        vq_copy_tile_bf16(Ll + aoff, Al, tid);
        vq_copy_tile_bf16(Ch + boff, Bh, tid);
        vq_copy_tile_bf16(Cl + boff, Bl, tid);
      }
#endif
      __syncthreads();   // tiles visible to all waves
      vq_mma_chunk(Ah, Al, Bh, Bl, w, lcol, lhalf, acc);
    }

    #pragma unroll
    for (int t = 0; t < 8; ++t) {
      const int colg = kBase + 16 * t + lcol;
      const float cn = cnorm[colg];
      #pragma unroll
      for (int r = 0; r < 8; ++r) {
        const float dist = cn - 2.0f * acc[t][r];
        if (dist < bestv[r]) { bestv[r] = dist; besti[r] = colg; }
      }
    }
  }

  #pragma unroll
  for (int r = 0; r < 8; ++r) {
    #pragma unroll
    for (int m = 8; m; m >>= 1) {
      const float ov = __shfl_xor(bestv[r], m, 32);
      const int   oi = __shfl_xor(besti[r], m, 32);
      if (ov < bestv[r] || (ov == bestv[r] && oi < besti[r])) {
        bestv[r] = ov; besti[r] = oi;
      }
    }
  }
  if (lcol == 0) {
    #pragma unroll
    for (int r = 0; r < 8; ++r) {
      const int rowg = rowBase + 16 * w + r + 8 * lhalf;
      pval[(size_t)rowg * 8 + blockIdx.y] = bestv[r];
      pidx[(size_t)rowg * 8 + blockIdx.y] = besti[r];
    }
  }
}

// ---------------------------------------------------------------------------
// Kernel 2b (fallback if workspace too small): fused fp32->bf16 staging
// ---------------------------------------------------------------------------
__device__ __forceinline__ void vq_stage128(const float* __restrict__ src,
                                            int rowBase, int colBase,
                                            __bf16* hi, __bf16* lo, int tid) {
  #pragma unroll
  for (int i = 0; i < 16; ++i) {
    const int f4 = tid + i * 256;
    const int r  = f4 >> 5;
    const int c4 = (f4 & 31) << 2;
    const float4 v = *(const float4*)(src + (size_t)(rowBase + r) * DIM + colBase + c4);
    float fx[4] = { v.x, v.y, v.z, v.w };
    v4bf h, lw;
    #pragma unroll
    for (int j = 0; j < 4; ++j) {
      const __bf16 hb = (__bf16)fx[j];
      h[j]  = hb;
      lw[j] = (__bf16)(fx[j] - (float)hb);
    }
    *(v4bf*)(hi + r * LDSS + c4) = h;
    *(v4bf*)(lo + r * LDSS + c4) = lw;
  }
}

__launch_bounds__(256)
__global__ void vq_dist_argmin_kernel(const float* __restrict__ latent,
                                      const float* __restrict__ codebook,
                                      const float* __restrict__ cnorm,
                                      float* __restrict__ pval,
                                      int*   __restrict__ pidx) {
  __shared__ __bf16 Ah[128 * LDSS];
  __shared__ __bf16 Al[128 * LDSS];
  __shared__ __bf16 Bh[128 * LDSS];
  __shared__ __bf16 Bl[128 * LDSS];

  const int tid   = threadIdx.x;
  const int w     = tid >> 5;
  const int l     = tid & 31;
  const int lhalf = l >> 4;
  const int lcol  = l & 15;
  const int rowBase    = blockIdx.x * 128;
  const int kSplitBase = blockIdx.y * 1024;

  float bestv[8];
  int   besti[8];
  #pragma unroll
  for (int r = 0; r < 8; ++r) { bestv[r] = 3.0e38f; besti[r] = 0; }

  const v8f vzero = {0.f, 0.f, 0.f, 0.f, 0.f, 0.f, 0.f, 0.f};

  for (int kt = 0; kt < 8; ++kt) {
    const int kBase = kSplitBase + kt * 128;
    v8f acc[8];
    #pragma unroll
    for (int t = 0; t < 8; ++t) acc[t] = vzero;

    for (int dc = 0; dc < 4; ++dc) {
      __syncthreads();
      vq_stage128(latent,   rowBase, dc * 128, Ah, Al, tid);
      vq_stage128(codebook, kBase,   dc * 128, Bh, Bl, tid);
      __syncthreads();
      vq_mma_chunk(Ah, Al, Bh, Bl, w, lcol, lhalf, acc);
    }

    #pragma unroll
    for (int t = 0; t < 8; ++t) {
      const int colg = kBase + 16 * t + lcol;
      const float cn = cnorm[colg];
      #pragma unroll
      for (int r = 0; r < 8; ++r) {
        const float dist = cn - 2.0f * acc[t][r];
        if (dist < bestv[r]) { bestv[r] = dist; besti[r] = colg; }
      }
    }
  }

  #pragma unroll
  for (int r = 0; r < 8; ++r) {
    #pragma unroll
    for (int m = 8; m; m >>= 1) {
      const float ov = __shfl_xor(bestv[r], m, 32);
      const int   oi = __shfl_xor(besti[r], m, 32);
      if (ov < bestv[r] || (ov == bestv[r] && oi < besti[r])) {
        bestv[r] = ov; besti[r] = oi;
      }
    }
  }
  if (lcol == 0) {
    #pragma unroll
    for (int r = 0; r < 8; ++r) {
      const int rowg = rowBase + 16 * w + r + 8 * lhalf;
      pval[(size_t)rowg * 8 + blockIdx.y] = bestv[r];
      pidx[(size_t)rowg * 8 + blockIdx.y] = besti[r];
    }
  }
}

// ---------------------------------------------------------------------------
// Kernel 3: final argmin over 8 K-split partials, gather codebook row,
// write quantized_st + index, per-row squared error. One wave per token.
// ---------------------------------------------------------------------------
__global__ void vq_assign_gather_kernel(const float* __restrict__ latent,
                                        const float* __restrict__ codebook,
                                        const float* __restrict__ pval,
                                        const int*   __restrict__ pidx,
                                        float* __restrict__ out_q,
                                        float* __restrict__ out_idx,
                                        float* __restrict__ rowsum) {
  const int w = threadIdx.x >> 5, l = threadIdx.x & 31;
  const int n = blockIdx.x * 8 + w;

  float bv = 3.0e38f;
  int   bi = 0x7fffffff;
  if (l < 8) { bv = pval[(size_t)n * 8 + l]; bi = pidx[(size_t)n * 8 + l]; }
  #pragma unroll
  for (int m = 4; m; m >>= 1) {
    const float ov = __shfl_xor(bv, m, 32);
    const int   oi = __shfl_xor(bi, m, 32);
    if (ov < bv || (ov == bv && oi < bi)) { bv = ov; bi = oi; }
  }
  bi = __shfl(bi, 0, 32);

  const float* crow = codebook + (size_t)bi * DIM;
  const float* xrow = latent   + (size_t)n  * DIM;
  float*       qrow = out_q    + (size_t)n  * DIM;
  float s = 0.f;
  for (int d = l; d < DIM; d += 32) {
    const float q = crow[d];
    const float x = xrow[d];
    qrow[d] = q;
    const float df = x - q;
    s += df * df;
  }
  #pragma unroll
  for (int m = 16; m; m >>= 1) s += __shfl_xor(s, m, 32);
  if (l == 0) { rowsum[n] = s; out_idx[n] = (float)bi; }
}

// ---------------------------------------------------------------------------
// Kernel 4: deterministic loss reduction: 1.25 * mean(diff^2)
// ---------------------------------------------------------------------------
__global__ void vq_loss_kernel(const float* __restrict__ rowsum,
                               float* __restrict__ out_loss) {
  __shared__ float sm[256];
  float s = 0.f;
  for (int i = threadIdx.x; i < N_TOK; i += 256) s += rowsum[i];
  sm[threadIdx.x] = s;
  __syncthreads();
  for (int off = 128; off; off >>= 1) {
    if ((int)threadIdx.x < off) sm[threadIdx.x] += sm[threadIdx.x + off];
    __syncthreads();
  }
  if (threadIdx.x == 0)
    *out_loss = 1.25f * sm[0] / ((float)N_TOK * (float)DIM);
}

// ---------------------------------------------------------------------------
extern "C" void kernel_launch(void* const* d_in, const int* in_sizes, int n_in,
                              void* d_out, int out_size, void* d_ws, size_t ws_size,
                              hipStream_t stream) {
  const float* latent   = (const float*)d_in[0];   // [8,1024,512]
  const float* codebook = (const float*)d_in[1];   // [8192,512]

  float* out      = (float*)d_out;
  float* out_q    = out;                              // 8*1024*512 floats
  float* out_loss = out + (size_t)N_TOK * DIM;        // 1 float
  float* out_idx  = out_loss + 1;                     // 8192 floats

  float* cnorm  = (float*)d_ws;                            // 8192 f
  float* pval   = cnorm + K_CB;                            // 8192*8 f
  int*   pidx   = (int*)(pval + (size_t)N_TOK * 8);        // 8192*8 i
  float* rowsum = (float*)(pidx + (size_t)N_TOK * 8);      // 8192 f

  const size_t small_bytes = (size_t)((char*)(rowsum + N_TOK) - (char*)d_ws);
  const size_t off_big     = (small_bytes + 255) & ~(size_t)255;
  const size_t arr_elems   = (size_t)N_TOK * DIM;          // == K_CB * DIM
  const size_t need_big    = off_big + 4 * arr_elems * 2;  // 4 bf16 arrays
  const bool   big         = ws_size >= need_big;          // constant per run

  __bf16* Lh = (__bf16*)((char*)d_ws + off_big);
  __bf16* Ll = Lh + arr_elems;
  __bf16* Ch = Ll + arr_elems;
  __bf16* Cl = Ch + arr_elems;

  vq_cnorm_kernel<<<K_CB / 8, 256, 0, stream>>>(codebook, cnorm);

  dim3 grid(64, 8);
  if (big) {
    const int nq = (int)(arr_elems / 4);                   // float4s per tensor
    vq_split_kernel<<<nq / 256, 256, 0, stream>>>(latent,   Lh, Ll);
    vq_split_kernel<<<nq / 256, 256, 0, stream>>>(codebook, Ch, Cl);
    vq_dist_argmin_bf16_kernel<<<grid, 256, 0, stream>>>(Lh, Ll, Ch, Cl,
                                                         cnorm, pval, pidx);
  } else {
    vq_dist_argmin_kernel<<<grid, 256, 0, stream>>>(latent, codebook,
                                                    cnorm, pval, pidx);
  }

  vq_assign_gather_kernel<<<N_TOK / 8, 256, 0, stream>>>(latent, codebook, pval, pidx,
                                                         out_q, out_idx, rowsum);

  vq_loss_kernel<<<1, 256, 0, stream>>>(rowsum, out_loss);
}
